// SupplyChainTemporalGNN_69612829933741
// MI455X (gfx1250) — compile-verified, matrix operand order checked by reading, not run on hardware
//
#include <hip/hip_runtime.h>

// Problem dims (fixed by reference)
#define B_    4
#define S_    16
#define N_    2048
#define F_    32
#define E_    16384
#define EF_   8
#define H_    64
#define L_    2
#define H2_   32
#define G_    (B_ * S_)        // 64 graphs
#define E2_   (E_ + N_)        // 18432 edges incl self loops
#define ROWS  (B_ * S_ * N_)   // 131072
#define TROWS (S_ * N_)        // 32768 rows per GRU step

typedef __attribute__((ext_vector_type(16))) _Float16 v16h;
typedef __attribute__((ext_vector_type(8)))  float    v8f;

__device__ __forceinline__ int kmap(int p) { return (p < 4) ? 2 * p : 16 + 2 * (p - 4); }

__device__ __forceinline__ v8f zero8() {
  v8f z = {0.f, 0.f, 0.f, 0.f, 0.f, 0.f, 0.f, 0.f};
  return z;
}

// A fragment (16x32 f16) from row-major f32 [*, lda], converting on the fly.
__device__ __forceinline__ v16h load_a_f32(const float* A, int lda, int row0, int k0, int lane) {
  const float* ap = A + (size_t)(row0 + (lane & 15)) * lda + k0 + ((lane >> 4) << 3);
  v16h a;
#pragma unroll
  for (int p = 0; p < 8; ++p) {
    int kb = kmap(p);
    a[2 * p]     = (_Float16)ap[kb];
    a[2 * p + 1] = (_Float16)ap[kb + 1];
  }
  return a;
}

// A fragment from row-major f16.
__device__ __forceinline__ v16h load_a_f16(const _Float16* A, int lda, int row0, int k0, int lane) {
  const _Float16* ap = A + (size_t)(row0 + (lane & 15)) * lda + k0 + ((lane >> 4) << 3);
  v16h a;
#pragma unroll
  for (int p = 0; p < 8; ++p) {
    int kb = kmap(p);
    a[2 * p]     = ap[kb];
    a[2 * p + 1] = ap[kb + 1];
  }
  return a;
}

// B fragment (32x16 f16): element (k, c) = Bm[k*bs_k + c*bs_n], f32 source.
__device__ __forceinline__ v16h load_b_f32(const float* Bm, int bs_k, int bs_n,
                                           int k0, int col0, int lane) {
  const float* bp = Bm + (size_t)(col0 + (lane & 15)) * bs_n +
                    (size_t)(k0 + ((lane >> 4) << 3)) * bs_k;
  v16h b;
#pragma unroll
  for (int p = 0; p < 8; ++p) {
    int kb = kmap(p);
    b[2 * p]     = (_Float16)bp[(size_t)kb * bs_k];
    b[2 * p + 1] = (_Float16)bp[(size_t)(kb + 1) * bs_k];
  }
  return b;
}

// ---------------- Weight pre-pack into WMMA B-fragment layout ------------------
// out layout: [nt][kc][lane] of v16h (32 bytes per lane-fragment)
__global__ void pack_b_kernel(const float* __restrict__ W, int bs_k, int bs_n,
                              int Kchunks, int Ntiles, _Float16* __restrict__ out) {
  int t = blockIdx.x * blockDim.x + threadIdx.x;
  if (t >= Ntiles * Kchunks * 32) return;
  int lane = t & 31;
  int idx = t >> 5;
  int kc = idx % Kchunks;
  int nt = idx / Kchunks;
  v16h b = load_b_f32(W, bs_k, bs_n, kc * 32, nt * 16, lane);
  ((v16h*)out)[t] = b;
}

// ---------------- Projection: X = x @ Wp + bp (f32 + f16 shadow) ---------------
__global__ void proj_kernel(const float* __restrict__ x, const v16h* __restrict__ BP,
                            const float* __restrict__ bias,
                            float* __restrict__ X, _Float16* __restrict__ XH) {
  int lane = threadIdx.x & 31;
  int wave = blockIdx.x * (blockDim.x >> 5) + (threadIdx.x >> 5);
  int row0 = wave * 16;
  if (row0 >= ROWS) return;
  v16h a = load_a_f32(x, F_, row0, 0, lane);
  int c = lane & 15;
  int rbase = row0 + ((lane >> 4) << 3);
#pragma unroll
  for (int nt = 0; nt < 4; ++nt) {
    v8f acc = __builtin_amdgcn_wmma_f32_16x16x32_f16(false, a, false, BP[nt * 32 + lane],
                                                     (short)0, zero8(), false, false);
    int col = nt * 16 + c;
    float bv = bias[col];
#pragma unroll
    for (int i = 0; i < 8; ++i) {
      float v = acc[i] + bv;
      size_t o = (size_t)(rbase + i) * H_ + col;
      X[o] = v;
      XH[o] = (_Float16)v;
    }
  }
}

// ---------------- Fused GAT linears: XL = A@Wl+bl, XR = A@Wr+br (A read once) --
__global__ void gat_linear_kernel(const _Float16* __restrict__ A,
                                  const v16h* __restrict__ B1, const float* __restrict__ bias1,
                                  const v16h* __restrict__ B2, const float* __restrict__ bias2,
                                  _Float16* __restrict__ C1, _Float16* __restrict__ C2) {
  int lane = threadIdx.x & 31;
  int wave = blockIdx.x * (blockDim.x >> 5) + (threadIdx.x >> 5);
  int row0 = wave * 16;
  if (row0 >= ROWS) return;
  v16h a0 = load_a_f16(A, H_, row0, 0, lane);
  v16h a1 = load_a_f16(A, H_, row0, 32, lane);
  int c = lane & 15;
  int rbase = row0 + ((lane >> 4) << 3);
#pragma unroll
  for (int nt = 0; nt < 4; ++nt) {
    int col = nt * 16 + c;
    v8f acc = __builtin_amdgcn_wmma_f32_16x16x32_f16(false, a0, false, B1[(nt * 2 + 0) * 32 + lane],
                                                     (short)0, zero8(), false, false);
    acc = __builtin_amdgcn_wmma_f32_16x16x32_f16(false, a1, false, B1[(nt * 2 + 1) * 32 + lane],
                                                 (short)0, acc, false, false);
    float bv1 = bias1[col];
#pragma unroll
    for (int i = 0; i < 8; ++i)
      C1[(size_t)(rbase + i) * H_ + col] = (_Float16)(acc[i] + bv1);
    acc = __builtin_amdgcn_wmma_f32_16x16x32_f16(false, a0, false, B2[(nt * 2 + 0) * 32 + lane],
                                                 (short)0, zero8(), false, false);
    acc = __builtin_amdgcn_wmma_f32_16x16x32_f16(false, a1, false, B2[(nt * 2 + 1) * 32 + lane],
                                                 (short)0, acc, false, false);
    float bv2 = bias2[col];
#pragma unroll
    for (int i = 0; i < 8; ++i)
      C2[(size_t)(rbase + i) * H_ + col] = (_Float16)(acc[i] + bv2);
  }
}

// ---------------- Graph preprocessing (deterministic, atomic-free) -------------
__global__ void loopattr_kernel(const int* __restrict__ dst, const float* __restrict__ edge_attr,
                                float* __restrict__ loop_attr, int* __restrict__ deg) {
  int n = blockIdx.x * blockDim.x + threadIdx.x;
  if (n >= N_) return;
  float acc[EF_];
#pragma unroll
  for (int f = 0; f < EF_; ++f) acc[f] = 0.f;
  int cnt = 0;
  for (int e = 0; e < E_; ++e) {
    if (dst[e] == n) {
      cnt++;
#pragma unroll
      for (int f = 0; f < EF_; ++f) acc[f] += edge_attr[(size_t)e * EF_ + f];
    }
  }
  float inv = 1.f / fmaxf((float)cnt, 1.f);
#pragma unroll
  for (int f = 0; f < EF_; ++f) loop_attr[(size_t)n * EF_ + f] = acc[f] * inv;
  deg[n] = cnt + 1;  // + self loop
}

__global__ void scan_kernel(const int* __restrict__ deg, int* __restrict__ row_ptr) {
  if (blockIdx.x == 0 && threadIdx.x == 0) {
    int s = 0;
    for (int n = 0; n < N_; ++n) { row_ptr[n] = s; s += deg[n]; }
    row_ptr[N_] = s;
  }
}

__global__ void csr_fill_kernel(const int* __restrict__ dst, const int* __restrict__ row_ptr,
                                int* __restrict__ col) {
  int n = blockIdx.x * blockDim.x + threadIdx.x;
  if (n >= N_) return;
  int o = row_ptr[n];
  for (int e = 0; e < E_; ++e)
    if (dst[e] == n) col[o++] = e;
  col[o] = E_ + n;  // self-loop edge id
}

// ---------------- ep = ea_f @ We[l] (K=8, tiny) --------------------------------
__global__ void ep_kernel(const float* __restrict__ edge_attr, const float* __restrict__ loop_attr,
                          const float* __restrict__ We_l, float* __restrict__ ep) {
  int idx = blockIdx.x * blockDim.x + threadIdx.x;
  if (idx >= E2_ * H_) return;
  int e = idx >> 6, h = idx & 63;
  const float* ea = (e < E_) ? (edge_attr + (size_t)e * EF_) : (loop_attr + (size_t)(e - E_) * EF_);
  float acc = 0.f;
#pragma unroll
  for (int f = 0; f < EF_; ++f) acc += ea[f] * We_l[f * H_ + h];
  ep[idx] = acc;
}

// ---------------- GATv2 attention logits (wave per (g,e)) ----------------------
__global__ void logits_kernel(const _Float16* __restrict__ XL, const _Float16* __restrict__ XR,
                              const float* __restrict__ ep, const float* __restrict__ att_l,
                              const int* __restrict__ src, const int* __restrict__ dst,
                              float* __restrict__ logit) {
  int lane = threadIdx.x & 31;
  long wave = (long)blockIdx.x * (blockDim.x >> 5) + (threadIdx.x >> 5);
  if (wave >= (long)G_ * E2_) return;
  int g = (int)(wave / E2_);
  int e = (int)(wave % E2_);
  int s, d;
  if (e < E_) { s = src[e]; d = dst[e]; } else { s = e - E_; d = s; }
  const _Float16* xl = XL + ((size_t)g * N_ + s) * H_;
  const _Float16* xr = XR + ((size_t)g * N_ + d) * H_;
  const float* epp = ep + (size_t)e * H_;
  float p = 0.f;
#pragma unroll
  for (int t = 0; t < 2; ++t) {
    int h = lane + t * 32;
    float m = (float)xl[h] + (float)xr[h] + epp[h];
    m = (m > 0.f) ? m : 0.2f * m;  // leaky_relu(0.2)
    p += m * att_l[h];
  }
#pragma unroll
  for (int off = 16; off > 0; off >>= 1) p += __shfl_xor(p, off);
  if (lane == 0) logit[wave] = p;
}

// ---------------- Segment softmax + aggregation (wave per (g,n)) ---------------
__global__ void aggregate_kernel(const _Float16* __restrict__ XL, const float* __restrict__ logit,
                                 const int* __restrict__ row_ptr, const int* __restrict__ col,
                                 const int* __restrict__ src, const float* __restrict__ gb_l,
                                 _Float16* __restrict__ out) {
  int lane = threadIdx.x & 31;
  int wave = blockIdx.x * (blockDim.x >> 5) + (threadIdx.x >> 5);
  if (wave >= G_ * N_) return;
  int g = wave >> 11;          // / N_
  int n = wave & (N_ - 1);
  int o0 = row_ptr[n], o1 = row_ptr[n + 1];
  const float* lg = logit + (size_t)g * E2_;
  float mx = -3.4e38f;
  for (int o = o0; o < o1; ++o) mx = fmaxf(mx, lg[col[o]]);
  float den = 0.f, a0 = 0.f, a1 = 0.f;
  for (int o = o0; o < o1; ++o) {
    int e = col[o];
    float p = expf(lg[e] - mx);
    den += p;
    int s = (e < E_) ? src[e] : (e - E_);
    const _Float16* xl = XL + ((size_t)g * N_ + s) * H_;
    a0 += p * (float)xl[lane];
    a1 += p * (float)xl[lane + 32];
  }
  float inv = 1.f / den;
  _Float16* op = out + ((size_t)g * N_ + n) * H_;
  op[lane]      = (_Float16)(a0 * inv + gb_l[lane]);
  op[lane + 32] = (_Float16)(a1 * inv + gb_l[lane + 32]);
}

// ---------------- Fused GRU step: gates GEMMs (WMMA) + nonlinearity ------------
// Xt: hg slice (f16). HprevH: f16 shadow of previous hidden (GEMM A). Hprev: f32
// previous hidden (state update). Writes Hnew f32 + HnewH f16 shadow.
__global__ void gru_step_kernel(const _Float16* __restrict__ Xt,
                                const _Float16* __restrict__ HprevH,
                                const float* __restrict__ Hprev,
                                float* __restrict__ Hnew, _Float16* __restrict__ HnewH,
                                const v16h* __restrict__ WihP,   // packed [12][2][32]
                                const v16h* __restrict__ WhhP,
                                const float* __restrict__ bih, const float* __restrict__ bhh) {
  int lane = threadIdx.x & 31;
  int wave = blockIdx.x * (blockDim.x >> 5) + (threadIdx.x >> 5);
  int mt = wave >> 2;   // 16-row tile
  int jt = wave & 3;    // 16-col tile within H
  int row0 = mt * 16;
  if (row0 >= TROWS) return;
  int j0 = jt * 16;
  bool hasH = (Hprev != nullptr);

  v16h ax[2], ah[2];
#pragma unroll
  for (int kc = 0; kc < 2; ++kc) ax[kc] = load_a_f16(Xt, H_, row0, kc * 32, lane);
  if (hasH)
#pragma unroll
    for (int kc = 0; kc < 2; ++kc) ah[kc] = load_a_f16(HprevH, H_, row0, kc * 32, lane);

  v8f gr = zero8(), gz = zero8(), gn = zero8();
  v8f hr = zero8(), hz = zero8(), hn = zero8();
#pragma unroll
  for (int kc = 0; kc < 2; ++kc) {
    v16h bR = WihP[((0 * 4 + jt) * 2 + kc) * 32 + lane];
    gr = __builtin_amdgcn_wmma_f32_16x16x32_f16(false, ax[kc], false, bR, (short)0, gr, false, false);
    v16h bZ = WihP[((1 * 4 + jt) * 2 + kc) * 32 + lane];
    gz = __builtin_amdgcn_wmma_f32_16x16x32_f16(false, ax[kc], false, bZ, (short)0, gz, false, false);
    v16h bN = WihP[((2 * 4 + jt) * 2 + kc) * 32 + lane];
    gn = __builtin_amdgcn_wmma_f32_16x16x32_f16(false, ax[kc], false, bN, (short)0, gn, false, false);
    if (hasH) {
      v16h cR = WhhP[((0 * 4 + jt) * 2 + kc) * 32 + lane];
      hr = __builtin_amdgcn_wmma_f32_16x16x32_f16(false, ah[kc], false, cR, (short)0, hr, false, false);
      v16h cZ = WhhP[((1 * 4 + jt) * 2 + kc) * 32 + lane];
      hz = __builtin_amdgcn_wmma_f32_16x16x32_f16(false, ah[kc], false, cZ, (short)0, hz, false, false);
      v16h cN = WhhP[((2 * 4 + jt) * 2 + kc) * 32 + lane];
      hn = __builtin_amdgcn_wmma_f32_16x16x32_f16(false, ah[kc], false, cN, (short)0, hn, false, false);
    }
  }

  int c = j0 + (lane & 15);
  int rbase = row0 + ((lane >> 4) << 3);
  float bir = bih[c], biz = bih[H_ + c], bin = bih[2 * H_ + c];
  float bhr = bhh[c], bhz = bhh[H_ + c], bhn = bhh[2 * H_ + c];
#pragma unroll
  for (int i = 0; i < 8; ++i) {
    size_t o = (size_t)(rbase + i) * H_ + c;
    float hp = hasH ? Hprev[o] : 0.f;
    float r  = 1.f / (1.f + expf(-(gr[i] + bir + hr[i] + bhr)));
    float z  = 1.f / (1.f + expf(-(gz[i] + biz + hz[i] + bhz)));
    float nn = tanhf(gn[i] + bin + r * (hn[i] + bhn));
    float hv = (1.f - z) * nn + z * hp;
    Hnew[o] = hv;
    HnewH[o] = (_Float16)hv;
  }
}

// ---------------- Output heads -------------------------------------------------
__global__ void head_kernel(const float* __restrict__ X,
                            const float* __restrict__ oW1, const float* __restrict__ ob1,
                            const float* __restrict__ oW2, const float* __restrict__ ob2,
                            const float* __restrict__ dW1, const float* __restrict__ db1,
                            const float* __restrict__ dW2, const float* __restrict__ db2,
                            float* __restrict__ out) {
  int r = blockIdx.x * blockDim.x + threadIdx.x;
  if (r >= B_ * N_) return;
  int b = r / N_, n = r % N_;
  const float* xl = X + ((size_t)b * S_ * N_ + (size_t)(S_ - 1) * N_ + n) * H_;
  float xo = ob2[0], xd = db2[0];
  for (int j = 0; j < H2_; ++j) {
    float ao = ob1[j], ad = db1[j];
#pragma unroll 8
    for (int k = 0; k < H_; ++k) {
      float xv = xl[k];
      ao += xv * oW1[k * H2_ + j];
      ad += xv * dW1[k * H2_ + j];
    }
    xo += fmaxf(ao, 0.f) * oW2[j];
    xd += fmaxf(ad, 0.f) * dW2[j];
  }
  out[r] = xo;
  out[B_ * N_ + r] = xd;
}

// ---------------- Host orchestration -------------------------------------------
extern "C" void kernel_launch(void* const* d_in, const int* in_sizes, int n_in,
                              void* d_out, int out_size, void* d_ws, size_t ws_size,
                              hipStream_t stream) {
  (void)in_sizes; (void)n_in; (void)out_size; (void)ws_size;
  const float* x        = (const float*)d_in[0];
  const int*   eidx     = (const int*)d_in[1];
  const int*   src      = eidx;
  const int*   dst      = eidx + E_;
  const float* edge_attr= (const float*)d_in[2];
  const float* Wp  = (const float*)d_in[3];
  const float* bp  = (const float*)d_in[4];
  const float* Wl  = (const float*)d_in[5];
  const float* bl  = (const float*)d_in[6];
  const float* Wr  = (const float*)d_in[7];
  const float* br  = (const float*)d_in[8];
  const float* We  = (const float*)d_in[9];
  const float* att = (const float*)d_in[10];
  const float* gb  = (const float*)d_in[11];
  const float* Wih = (const float*)d_in[12];
  const float* Whh = (const float*)d_in[13];
  const float* bih = (const float*)d_in[14];
  const float* bhh = (const float*)d_in[15];
  const float* oW1 = (const float*)d_in[16];
  const float* ob1 = (const float*)d_in[17];
  const float* oW2 = (const float*)d_in[18];
  const float* ob2 = (const float*)d_in[19];
  const float* dW1 = (const float*)d_in[20];
  const float* db1 = (const float*)d_in[21];
  const float* dW2 = (const float*)d_in[22];
  const float* db2 = (const float*)d_in[23];

  char* ws = (char*)d_ws;
  size_t off = 0;
  auto alloc = [&](size_t bytes) -> void* {
    void* p = ws + off;
    off = (off + bytes + 255) & ~(size_t)255;
    return p;
  };
  float*    X      = (float*)alloc((size_t)ROWS * H_ * 4);
  _Float16* XH     = (_Float16*)alloc((size_t)ROWS * H_ * 2);   // f16 shadow of X
  _Float16* HG     = (_Float16*)alloc((size_t)ROWS * H_ * 2);   // GAT output (GRU input)
  _Float16* XLb    = (_Float16*)alloc((size_t)ROWS * H_ * 2);
  _Float16* XRb    = (_Float16*)alloc((size_t)ROWS * H_ * 2);
  float*    EP     = (float*)alloc((size_t)E2_ * H_ * 4);
  float*    LOGIT  = (float*)alloc((size_t)G_ * E2_ * 4);
  float*    LOOPA  = (float*)alloc((size_t)N_ * EF_ * 4);
  int*      DEG    = (int*)alloc((size_t)N_ * 4);
  int*      ROWPTR = (int*)alloc((size_t)(N_ + 1) * 4);
  int*      COL    = (int*)alloc((size_t)E2_ * 4);
  // Packed weight fragments (f16, 32B per lane-fragment)
  const size_t FR = 32 * 16;  // elements per (nt,kc) fragment group
  _Float16* WpP  = (_Float16*)alloc(4 * 1 * FR * 2);
  _Float16* WlP  = (_Float16*)alloc((size_t)L_ * 4 * 2 * FR * 2);
  _Float16* WrP  = (_Float16*)alloc((size_t)L_ * 4 * 2 * FR * 2);
  _Float16* WihP = (_Float16*)alloc((size_t)L_ * 12 * 2 * FR * 2);
  _Float16* WhhP = (_Float16*)alloc((size_t)L_ * 12 * 2 * FR * 2);

  // Graph preprocessing: self-loop attrs + CSR over dst (deterministic)
  loopattr_kernel<<<(N_ + 255) / 256, 256, 0, stream>>>(dst, edge_attr, LOOPA, DEG);
  scan_kernel<<<1, 32, 0, stream>>>(DEG, ROWPTR);
  csr_fill_kernel<<<(N_ + 255) / 256, 256, 0, stream>>>(dst, ROWPTR, COL);

  // Pack all weights into WMMA fragment layout (tiny, L2-resident)
  pack_b_kernel<<<1, 128, 0, stream>>>(Wp, H_, 1, 1, 4, WpP);
  for (int l = 0; l < L_; ++l) {
    pack_b_kernel<<<1, 256, 0, stream>>>(Wl + (size_t)l * H_ * H_, H_, 1, 2, 4, WlP + (size_t)l * 4 * 2 * FR);
    pack_b_kernel<<<1, 256, 0, stream>>>(Wr + (size_t)l * H_ * H_, H_, 1, 2, 4, WrP + (size_t)l * 4 * 2 * FR);
    // gi = xt @ Wih.T -> B(k,c) = Wih[c*H + k]  (bs_k=1, bs_n=H), 12 col-tiles
    pack_b_kernel<<<3, 256, 0, stream>>>(Wih + (size_t)l * 3 * H_ * H_, 1, H_, 2, 12, WihP + (size_t)l * 12 * 2 * FR);
    pack_b_kernel<<<3, 256, 0, stream>>>(Whh + (size_t)l * 3 * H_ * H_, 1, H_, 2, 12, WhhP + (size_t)l * 12 * 2 * FR);
  }

  const int gemm_blocks = ((ROWS / 16) * 32 + 255) / 256;  // wave per 16-row tile
  proj_kernel<<<gemm_blocks, 256, 0, stream>>>(x, (const v16h*)WpP, bp, X, XH);

  for (int l = 0; l < L_; ++l) {
    ep_kernel<<<(E2_ * H_ + 255) / 256, 256, 0, stream>>>(
        edge_attr, LOOPA, We + (size_t)l * EF_ * H_, EP);
    gat_linear_kernel<<<gemm_blocks, 256, 0, stream>>>(
        XH, (const v16h*)(WlP + (size_t)l * 4 * 2 * FR), bl + l * H_,
        (const v16h*)(WrP + (size_t)l * 4 * 2 * FR), br + l * H_, XLb, XRb);
    long lwaves = (long)G_ * E2_;
    logits_kernel<<<(int)((lwaves * 32 + 255) / 256), 256, 0, stream>>>(
        XLb, XRb, EP, att + l * H_, src, dst, LOGIT);
    aggregate_kernel<<<((G_ * N_) * 32 + 255) / 256, 256, 0, stream>>>(
        XLb, LOGIT, ROWPTR, COL, src, gb + l * H_, HG);
    // GRU scan over batch axis (4 steps); updates X (f32) + XH (f16) in place
    for (int t = 0; t < B_; ++t) {
      const _Float16* Xt = HG + (size_t)t * TROWS * H_;
      const _Float16* HpH = (t == 0) ? nullptr : (XH + (size_t)(t - 1) * TROWS * H_);
      const float* Hp = (t == 0) ? nullptr : (X + (size_t)(t - 1) * TROWS * H_);
      float* Hn = X + (size_t)t * TROWS * H_;
      _Float16* HnH = XH + (size_t)t * TROWS * H_;
      int gwaves = (TROWS / 16) * 4;
      gru_step_kernel<<<(gwaves * 32 + 255) / 256, 256, 0, stream>>>(
          Xt, HpH, Hp, Hn, HnH,
          (const v16h*)(WihP + (size_t)l * 12 * 2 * FR),
          (const v16h*)(WhhP + (size_t)l * 12 * 2 * FR),
          bih + (size_t)l * 3 * H_, bhh + (size_t)l * 3 * H_);
    }
  }
  head_kernel<<<(B_ * N_ + 255) / 256, 256, 0, stream>>>(
      X, oW1, ob1, oW2, ob2, dW1, db1, dW2, db2, (float*)d_out);
}
